// CrossLayerSpatialWiseFrequencyAttention_16870631539405
// MI455X (gfx1250) — compile-verified, matrix operand order checked by reading, not run on hardware
//
#include <hip/hip_runtime.h>
#include <hip/hip_bf16.h>

// ---------------------------------------------------------------------------
// CrossLayerSpatialWiseFrequencyAttention for MI455X (gfx1250, wave32, WMMA)
// B=8, C=256, NH=4, HD=64, 20x20=400 windows/level, 25+9+1=35 tokens/window.
// All GEMMs on v_wmma_f32_16x16x32_f16 (f16 in, f32 accumulate).
// q/k/v attention tiles staged into LDS via the Tensor Data Mover when the
// toolchain exposes __builtin_amdgcn_tensor_load_to_lds (TENSORcnt waits),
// otherwise via b128 cooperative loads.
// ---------------------------------------------------------------------------

typedef __attribute__((ext_vector_type(16))) _Float16 v16h;
typedef __attribute__((ext_vector_type(8)))  _Float16 v8h;
typedef __attribute__((ext_vector_type(8)))  float    v8f;

__device__ __forceinline__ v8f wmma_f16(v16h a, v16h b, v8f c) {
  return __builtin_amdgcn_wmma_f32_16x16x32_f16(false, a, false, b, (short)0, c,
                                                false, false);
}

#define SHUF16(lo, hi) \
  __builtin_shufflevector(lo, hi, 0, 1, 2, 3, 4, 5, 6, 7, 8, 9, 10, 11, 12, 13, 14, 15)

#define NB    8
#define NC    256
#define NHD   4
#define HD    64
#define NWIN  400
#define TT    35
#define TBL   81

// output bases (fp32 elements)
#define OUT0_BASE 0L
#define OUT1_BASE 13107200L            // 8*256*80*80
#define OUT2_BASE 16384000L            // + 8*256*40*40
#define OUT_N     17203200L            // + 8*256*20*20

// workspace layout (byte offsets, 16B aligned)
#define OFF_Q      0L
#define OFF_K      57344000L
#define OFF_V      114688000L
#define OFF_CTX    172032000L          // [b][w][t][NH*HD] f16
#define OFF_POS    229376000L          // [NH][35][35] f32
#define OFF_WQ0    229395600L          // transposed [768][256] f16 per level
#define OFF_WQ1    229788816L
#define OFF_WQ2    230182032L
#define OFF_WP     230575248L          // transposed [256][256] f16

// ---------------------------------------------------------------------------
// Tensor Data Mover: 1-D contiguous tile load (global -> LDS), D# per ISA ch8.
// This toolchain exposes the 6-arg builtin:
//   (uint32x4 g0, int32x8 g1, int32x4 g2, int32x4 g3, int32x8 extra, i32 cpol)
#if __has_builtin(__builtin_amdgcn_tensor_load_to_lds)
#define HAVE_TDM 1
typedef __attribute__((ext_vector_type(4))) unsigned int u32x4;
typedef __attribute__((ext_vector_type(8))) int i32x8;
typedef __attribute__((ext_vector_type(4))) int i32x4;

__device__ __forceinline__ void tdm_load_1d(const void* gsrc, const void* ldsdst,
                                            unsigned n8 /* 8-byte elements */) {
  const unsigned long long ga = (unsigned long long)gsrc;
  const unsigned lds = (unsigned)(unsigned long long)ldsdst;
  u32x4 g0;
  g0[0] = 1u;                                          // count=1 (valid user D#)
  g0[1] = lds;                                         // lds_addr
  g0[2] = (unsigned)(ga & 0xffffffffu);                // global_addr[31:0]
  g0[3] = (unsigned)((ga >> 32) & 0x01ffffffu) | (2u << 30); // addr[56:32], type=2
  i32x8 g1;
  g1[0] = (3 << 16);                                   // data_size=3 (8B), no mcast
  g1[1] = (int)((n8 & 0xffffu) << 16);                 // tensor_dim0[15:0]
  g1[2] = (int)((n8 >> 16) & 0xffffu) | (1 << 16);     // tensor_dim0[31:16], dim1=1
  g1[3] = (int)((n8 & 0xffffu) << 16);                 // tile_dim0 = n8
  g1[4] = 0;                                           // tile_dim1/2 unused
  g1[5] = (int)n8;                                     // tensor_dim0_stride[31:0]
  g1[6] = 0;
  g1[7] = 0;
  i32x4 z4 = {};
  i32x8 z8 = {};
  __builtin_amdgcn_tensor_load_to_lds(g0, g1, z4, z4, z8, 0);
}
#endif

// ---------------------------------------------------------------------------
__global__ void zero_kernel(float* __restrict__ p, long n) {
  long i = (long)blockIdx.x * blockDim.x + threadIdx.x;
  if (i < n) p[i] = 0.0f;
}

// ---------------------------------------------------------------------------
// convert weights f32 -> f16 AND transpose to [n][k] so GEMM B-fragments are
// contiguous along k (two b128 loads per fragment).
__global__ void cvt_weights_kernel(const float* __restrict__ w0,
                                   const float* __restrict__ w1,
                                   const float* __restrict__ w2,
                                   const float* __restrict__ wp,
                                   _Float16* __restrict__ wq0t,
                                   _Float16* __restrict__ wq1t,
                                   _Float16* __restrict__ wq2t,
                                   _Float16* __restrict__ wpt) {
  const int NW = 256 * 768;
  int i = blockIdx.x * blockDim.x + threadIdx.x;
  if (i < 3 * NW) {
    const int l = i / NW, j = i % NW;
    const int n = j >> 8, k = j & 255;                 // out[n][k] = in[k][n]
    const float* src = (l == 0) ? w0 : (l == 1) ? w1 : w2;
    _Float16* dst = (l == 0) ? wq0t : (l == 1) ? wq1t : wq2t;
    dst[j] = (_Float16)src[k * 768 + n];
  } else if (i < 3 * NW + 65536) {
    const int j = i - 3 * NW;
    const int n = j >> 8, k = j & 255;
    wpt[j] = (_Float16)wp[k * 256 + n];
  }
}

// ---------------------------------------------------------------------------
// relative-position bias pos[h][q][k] (REL_IDX recomputed on device)
__device__ __forceinline__ void tok_coord(int t, float* ch, float* cw, int* lvl) {
  if (t < 25)      { int i = t / 5, j = t % 5; *ch = (float)(i - 2); *cw = (float)(j - 2); *lvl = 0; }
  else if (t < 34) { int u = t - 25; int i = u / 3, j = u % 3; const float s = 5.0f / 3.0f;
                     *ch = (float)(i - 1) * s; *cw = (float)(j - 1) * s; *lvl = 1; }
  else             { *ch = 0.0f; *cw = 0.0f; *lvl = 2; }
}

__global__ void pos_kernel(const float* __restrict__ rpb,
                           const float* __restrict__ abias,
                           float* __restrict__ pos) {
  int i = blockIdx.x * blockDim.x + threadIdx.x;
  if (i >= TT * TT) return;
  int q = i / TT, k = i % TT;
  float qh, qw, kh, kw; int ql, kl;
  tok_coord(q, &qh, &qw, &ql);
  tok_coord(k, &kh, &kw, &kl);
  float idx = 9.0f * (qh - kh + 4.0f) + (qw - kw + 4.0f);
  int fi = (int)floorf(idx); fi = fi < 0 ? 0 : (fi > TBL - 1 ? TBL - 1 : fi);
  int ci = (int)ceilf(idx);  ci = ci < 0 ? 0 : (ci > TBL - 1 ? TBL - 1 : ci);
  float wce = idx - (float)fi;
  #pragma unroll
  for (int h = 0; h < NHD; ++h) {
    pos[((long)h * TT + q) * TT + k] =
        (1.0f - wce) * rpb[fi * NHD + h] + wce * rpb[ci * NHD + h]
      + abias[ql * NHD + h];
  }
}

// ---------------------------------------------------------------------------
// fused unfold + QKV projection. Block = 4 waves = one 16-row tile shared via
// LDS across 4 column tiles. A loads are unconditional (mask-multiply, no
// exec-mask dances); fragments come from LDS as b128 reads.
#define SA_STRIDE 264                  // 256 + 8 halves: kills LDS bank conflicts
__global__ __launch_bounds__(128) void qkv_gemm_kernel(
    const float* __restrict__ x, const _Float16* __restrict__ wqt,
    _Float16* __restrict__ qd, _Float16* __restrict__ kd, _Float16* __restrict__ vd,
    int Hl, int ksz, int stride, int pad, int T, int tok_base) {
  __shared__ __align__(16) _Float16 sA[16 * SA_STRIDE];

  const int tid  = threadIdx.x;
  const int lane = tid & 31;
  const int wv   = tid >> 5;
  const int rowTile = blockIdx.y;
  const int n    = (blockIdx.x * 4 + wv) * 16 + (lane & 15);   // 0..767
  const long HW  = (long)Hl * Hl;

  // ---- stage A tile (16 tokens x 256 channels) into LDS, coalesced --------
  {
    const int row = tid & 15;
    const int m   = rowTile * 16 + row;
    const int t   = m % T;
    const int win = (m / T) % NWIN;
    const int b   = m / (T * NWIN);
    const int hh  = (win / 20) * stride - pad + (t / ksz);
    const int ww  = (win % 20) * stride - pad + (t % ksz);
    const bool inb = (hh >= 0) && (hh < Hl) && (ww >= 0) && (ww < Hl);
    const float msk = inb ? 1.0f : 0.0f;
    const float* xs = x + (long)b * NC * HW + (inb ? ((long)hh * Hl + ww) : 0L);
    for (int c = (tid >> 4); c < NC; c += 8)
      sA[row * SA_STRIDE + c] = (_Float16)(msk * xs[(long)c * HW]);
  }
  __syncthreads();

  // ---- 16x16 tile GEMM, K=256 in 8 WMMA steps -----------------------------
  const v8h* arow = (const v8h*)(sA + (lane & 15) * SA_STRIDE);
  const v8h* brow = (const v8h*)(wqt + (long)n * NC);
  const int  k8   = (lane >= 16) ? 1 : 0;
  __builtin_prefetch(brow, 0, 1);

  v8f acc = {};
  #pragma unroll
  for (int kk = 0; kk < 8; ++kk) {
    v16h a  = SHUF16(arow[kk * 4 + k8], arow[kk * 4 + k8 + 2]);
    v16h bf = SHUF16(brow[kk * 4 + k8], brow[kk * 4 + k8 + 2]);
    acc = wmma_f16(a, bf, acc);
  }

  // ---- scatter to q/k/v:  n = s*256 + head*64 + d -------------------------
  const int s  = n >> 8;
  const int hd = (n >> 6) & 3;
  const int d  = n & 63;
  _Float16* dst = (s == 0) ? qd : (s == 1) ? kd : vd;
  #pragma unroll
  for (int r = 0; r < 8; ++r) {
    const int mr = rowTile * 16 + r + ((lane >= 16) ? 8 : 0);
    const int tr = mr % T;
    const int wr = (mr / T) % NWIN;
    const int br = mr / (T * NWIN);
    const long o = ((((long)br * NHD + hd) * NWIN + wr) * TT + (tok_base + tr)) * HD + d;
    dst[o] = (_Float16)acc[r];
  }
}

// ---------------------------------------------------------------------------
// attention: one workgroup (9 waves, 288 threads) per (b, head, window)
__global__ __launch_bounds__(288) void attn_kernel(
    const _Float16* __restrict__ qg, const _Float16* __restrict__ kg,
    const _Float16* __restrict__ vg, const float* __restrict__ pos,
    _Float16* __restrict__ ctx) {
  __shared__ __align__(16) _Float16 qn[48 * 64];
  __shared__ __align__(16) _Float16 kn[48 * 64];
  __shared__ __align__(16) _Float16 vstage[TT * 64];   // raw v rows
  __shared__ __align__(16) _Float16 vT[64 * 64];       // v transposed [d][k]
  __shared__ float    sS[48 * 48];
  __shared__ __align__(16) _Float16 pP[48 * 64];

  const int tid  = threadIdx.x;
  const int lane = tid & 31;
  const int wv   = tid >> 5;                           // 0..8
  const int idx  = blockIdx.x;
  const int w    = idx % NWIN;
  const int h    = (idx / NWIN) & 3;
  const int b    = idx / (NWIN * NHD);
  const long base = ((((long)b * NHD + h) * NWIN + w) * TT) * HD;
  const v8h zz = {};

#ifdef HAVE_TDM
  // ---- TDM: three contiguous 4480B tiles, TENSORcnt-tracked ---------------
  if (wv == 0) {
    tdm_load_1d(qg + base, qn,     (TT * HD * 2) / 8);
    tdm_load_1d(kg + base, kn,     (TT * HD * 2) / 8);
    tdm_load_1d(vg + base, vstage, (TT * HD * 2) / 8);
    __builtin_amdgcn_s_wait_tensorcnt(0);
  }
  for (int i = tid; i < 13 * 8; i += 288) {            // zero pad rows 35..47
    ((v8h*)(qn + TT * HD))[i] = zz;
    ((v8h*)(kn + TT * HD))[i] = zz;
  }
#else
  const v8h* qsv = (const v8h*)(qg + base);
  const v8h* ksv = (const v8h*)(kg + base);
  const v8h* vsv = (const v8h*)(vg + base);
  for (int i = tid; i < 48 * 8; i += 288) {
    const int r = i >> 3;
    ((v8h*)qn)[i] = (r < TT) ? qsv[i] : zz;
    ((v8h*)kn)[i] = (r < TT) ? ksv[i] : zz;
  }
  for (int i = tid; i < TT * 8; i += 288) ((v8h*)vstage)[i] = vsv[i];
#endif
  __syncthreads();

  // ---- transpose v into vT[d][k] (zero-padded K), L2-normalize q/k rows ---
  for (int i = tid; i < 64 * 64; i += 288) {
    const int d = i >> 6, k = i & 63;
    vT[i] = (k < TT) ? vstage[k * HD + d] : (_Float16)0.0f;
  }
  if (tid < TT) {
    float ss = 0.0f;
    for (int c = 0; c < HD; ++c) { float v = (float)qn[tid * HD + c]; ss += v * v; }
    const float sc = 1.0f / fmaxf(sqrtf(ss), 1e-12f);
    for (int c = 0; c < HD; ++c)
      qn[tid * HD + c] = (_Float16)((float)qn[tid * HD + c] * sc);
  } else if (tid >= 64 && tid < 64 + TT) {
    const int r = tid - 64;
    float ss = 0.0f;
    for (int c = 0; c < HD; ++c) { float v = (float)kn[r * HD + c]; ss += v * v; }
    const float sc = 1.0f / fmaxf(sqrtf(ss), 1e-12f);
    for (int c = 0; c < HD; ++c)
      kn[r * HD + c] = (_Float16)((float)kn[r * HD + c] * sc);
  }
  __syncthreads();

  // ---- S = qn * kn^T (3x3 tiles, one per wave) + pos, mask ----------------
  {
    const int tm = wv / 3, tn = wv % 3;
    const int k8 = (lane >= 16) ? 1 : 0;
    const v8h* qrow = (const v8h*)(qn + (tm * 16 + (lane & 15)) * HD);
    const v8h* krow = (const v8h*)(kn + (tn * 16 + (lane & 15)) * HD);
    v8f acc = {};
    #pragma unroll
    for (int kk = 0; kk < 2; ++kk) {
      v16h a  = SHUF16(qrow[kk * 4 + k8], qrow[kk * 4 + k8 + 2]);
      v16h bb = SHUF16(krow[kk * 4 + k8], krow[kk * 4 + k8 + 2]);
      acc = wmma_f16(a, bb, acc);
    }
    const int nn = tn * 16 + (lane & 15);
    #pragma unroll
    for (int r = 0; r < 8; ++r) {
      const int mm = tm * 16 + r + ((lane >= 16) ? 8 : 0);
      float v = acc[r];
      if (nn < TT && mm < TT) v += pos[((long)h * TT + mm) * TT + nn];
      if (nn >= TT) v = -1e30f;
      sS[mm * 48 + nn] = v;
    }
  }
  __syncthreads();

  // ---- per-row softmax -> P (f16, zero padded to 48x64) -------------------
  if (tid < 48) {
    const int r = tid;
    if (r < TT) {
      float mx = -1e30f;
      for (int c = 0; c < TT; ++c) mx = fmaxf(mx, sS[r * 48 + c]);
      float sum = 0.0f;
      for (int c = 0; c < TT; ++c) { float e = __expf(sS[r * 48 + c] - mx); sS[r * 48 + c] = e; sum += e; }
      const float inv = 1.0f / sum;
      for (int c = 0; c < 64; ++c)
        pP[r * 64 + c] = (c < TT) ? (_Float16)(sS[r * 48 + c] * inv) : (_Float16)0.0f;
    } else {
      for (int c = 0; c < 64; ++c) pP[r * 64 + c] = (_Float16)0.0f;
    }
  }
  __syncthreads();

  // ---- O = P * V (48x64, 3x4 tiles over 9 waves) --------------------------
  for (int tile = wv; tile < 12; tile += 9) {
    const int tm = tile >> 2, tn = tile & 3;
    const int k8 = (lane >= 16) ? 1 : 0;
    const v8h* prow = (const v8h*)(pP + (tm * 16 + (lane & 15)) * 64);
    const v8h* vrow = (const v8h*)(vT + (tn * 16 + (lane & 15)) * 64);
    v8f acc = {};
    #pragma unroll
    for (int kk = 0; kk < 2; ++kk) {
      v16h a  = SHUF16(prow[kk * 4 + k8], prow[kk * 4 + k8 + 2]);
      v16h bb = SHUF16(vrow[kk * 4 + k8], vrow[kk * 4 + k8 + 2]);
      acc = wmma_f16(a, bb, acc);
    }
    const int d = tn * 16 + (lane & 15);
    #pragma unroll
    for (int r = 0; r < 8; ++r) {
      const int mm = tm * 16 + r + ((lane >= 16) ? 8 : 0);
      if (mm < TT) {
        const long o = (((long)b * NWIN + w) * TT + mm) * NC + h * HD + d;
        ctx[o] = (_Float16)acc[r];
      }
    }
  }
}

// ---------------------------------------------------------------------------
// output projection [112000 x 256] x [256 x 256] + bias, then fold (atomicAdd)
__global__ __launch_bounds__(128) void proj_fold_kernel(
    const _Float16* __restrict__ ctx, const _Float16* __restrict__ wpt,
    const float* __restrict__ bias, float* __restrict__ out) {
  const int lane = threadIdx.x & 31;
  const int wv   = threadIdx.x >> 5;
  const int rowTile = blockIdx.y * 4 + wv;
  const int n    = blockIdx.x * 16 + (lane & 15);      // channel 0..255
  const int m    = rowTile * 16 + (lane & 15);
  const int k8   = (lane >= 16) ? 1 : 0;

  const v8h* arow = (const v8h*)(ctx + (long)m * NC);
  const v8h* brow = (const v8h*)(wpt + (long)n * NC);
  __builtin_prefetch(brow, 0, 1);

  v8f acc = {};
  #pragma unroll
  for (int kk = 0; kk < 8; ++kk) {
    v16h a  = SHUF16(arow[kk * 4 + k8], arow[kk * 4 + k8 + 2]);
    v16h bf = SHUF16(brow[kk * 4 + k8], brow[kk * 4 + k8 + 2]);
    acc = wmma_f16(a, bf, acc);
  }

  const float bn = bias[n];
  #pragma unroll
  for (int r = 0; r < 8; ++r) {
    const int mr  = rowTile * 16 + r + ((lane >= 16) ? 8 : 0);
    const float v = acc[r] + bn;
    const int t35 = mr % TT;
    const int win = (mr / TT) % NWIN;
    const int b   = mr / (TT * NWIN);
    int ksz, stride, pad, Hl, tt; long lbase;
    if (t35 < 25)      { ksz = 5; stride = 4; pad = 2; Hl = 80; tt = t35;      lbase = OUT0_BASE; }
    else if (t35 < 34) { ksz = 3; stride = 2; pad = 1; Hl = 40; tt = t35 - 25; lbase = OUT1_BASE; }
    else               { ksz = 1; stride = 1; pad = 0; Hl = 20; tt = 0;        lbase = OUT2_BASE; }
    const int hh = (win / 20) * stride - pad + (tt / ksz);
    const int ww = (win % 20) * stride - pad + (tt % ksz);
    if (hh >= 0 && hh < Hl && ww >= 0 && ww < Hl) {
      const long o = lbase + ((long)b * NC + n) * Hl * Hl + (long)hh * Hl + ww;
      atomicAdd(&out[o], v);
    }
  }
}

// ---------------------------------------------------------------------------
extern "C" void kernel_launch(void* const* d_in, const int* in_sizes, int n_in,
                              void* d_out, int out_size, void* d_ws, size_t ws_size,
                              hipStream_t stream) {
  const float* x0    = (const float*)d_in[0];
  const float* x1    = (const float*)d_in[1];
  const float* x2    = (const float*)d_in[2];
  const float* wqkv0 = (const float*)d_in[3];
  const float* wqkv1 = (const float*)d_in[4];
  const float* wqkv2 = (const float*)d_in[5];
  const float* wproj = (const float*)d_in[6];
  const float* bproj = (const float*)d_in[7];
  const float* rpb   = (const float*)d_in[8];
  const float* abias = (const float*)d_in[9];
  float* out = (float*)d_out;

  char* ws = (char*)d_ws;
  _Float16* qh   = (_Float16*)(ws + OFF_Q);
  _Float16* kh   = (_Float16*)(ws + OFF_K);
  _Float16* vh   = (_Float16*)(ws + OFF_V);
  _Float16* ctxh = (_Float16*)(ws + OFF_CTX);
  float*    posf = (float*)   (ws + OFF_POS);
  _Float16* wq0t = (_Float16*)(ws + OFF_WQ0);
  _Float16* wq1t = (_Float16*)(ws + OFF_WQ1);
  _Float16* wq2t = (_Float16*)(ws + OFF_WQ2);
  _Float16* wpt  = (_Float16*)(ws + OFF_WP);

  // 1) zero output (fold is overlap-add)
  zero_kernel<<<(unsigned)((OUT_N + 255) / 256), 256, 0, stream>>>(out, OUT_N);

  // 2) weights -> f16, transposed
  const int cvtN = 3 * 256 * 768 + 256 * 256;
  cvt_weights_kernel<<<(cvtN + 255) / 256, 256, 0, stream>>>(
      wqkv0, wqkv1, wqkv2, wproj, wq0t, wq1t, wq2t, wpt);

  // 3) relative-position bias table
  pos_kernel<<<(TT * TT + 255) / 256, 256, 0, stream>>>(rpb, abias, posf);

  // 4) fused unfold + QKV GEMMs (rowTiles: 5000/1800/200; 12 x-blocks of 4 col tiles)
  qkv_gemm_kernel<<<dim3(12, 5000), 128, 0, stream>>>(x0, wq0t, qh, kh, vh, 80, 5, 4, 2, 25, 0);
  qkv_gemm_kernel<<<dim3(12, 1800), 128, 0, stream>>>(x1, wq1t, qh, kh, vh, 40, 3, 2, 1,  9, 25);
  qkv_gemm_kernel<<<dim3(12,  200), 128, 0, stream>>>(x2, wq2t, qh, kh, vh, 20, 1, 1, 0,  1, 34);

  // 5) attention: one WG per (b, head, window)
  attn_kernel<<<NB * NHD * NWIN, 288, 0, stream>>>(qh, kh, vh, posf, ctxh);

  // 6) projection + fold scatter (rowTiles: 7000; 16 col tiles)
  proj_fold_kernel<<<dim3(16, 1750), 128, 0, stream>>>(ctxh, wpt, bproj, out);
}
